// LDConv_9526237462558
// MI455X (gfx1250) — matrix operand order repaired and checked by reading, not compiled, not run on hardware
//
#include <hip/hip_runtime.h>

typedef __bf16 v16bf __attribute__((ext_vector_type(16)));
typedef float  v8f   __attribute__((ext_vector_type(8)));

#define H 80
#define W 80
#define HW 6400
#define IN_C 128
#define OUT_C 256
#define NPAR 9
#define KDIM 1152          // IN_C * NPAR
#define KT_STEPS 36        // KDIM / 32
#define BATCH 8
#define BN_EPS 1e-5f

// ---------------------------------------------------------------------------
// Kernel 1: offset-predicting 3x3 conv (128 -> 18 channels, SAME padding)
// offs layout: [b][co(18)][h*w]
// ---------------------------------------------------------------------------
__global__ __launch_bounds__(256) void ldconv_offset(
    const float* __restrict__ x, const float* __restrict__ wp,
    const float* __restrict__ bp, float* __restrict__ offs)
{
    int t = blockIdx.x * 256 + threadIdx.x;          // 0 .. 8*18*6400-1
    int pix = t % HW;
    int tmp = t / HW;
    int co  = tmp % 18;
    int b   = tmp / 18;
    int h = pix / W, w = pix % W;

    float acc = bp[co];
    const float* xb = x  + (size_t)b * IN_C * HW;
    const float* wc = wp + (size_t)co * IN_C * 9;
    for (int ic = 0; ic < IN_C; ++ic) {
        const float* xp = xb + (size_t)ic * HW;
        const float* wk = wc + ic * 9;
        #pragma unroll
        for (int ky = 0; ky < 3; ++ky) {
            int hy = h + ky - 1;
            if (hy < 0 || hy >= H) continue;
            #pragma unroll
            for (int kx = 0; kx < 3; ++kx) {
                int wx = w + kx - 1;
                if (wx < 0 || wx >= W) continue;
                acc += wk[ky * 3 + kx] * xp[hy * W + wx];
            }
        }
    }
    offs[t] = acc;
}

// ---------------------------------------------------------------------------
// Kernel 2: pack w_conv (256,128,9,1) fp32 -> bf16 B-fragments.
// Fragment layout (contiguous k order, matches A staging) 16-bit B 32x16: lane g=lane/16, n=lane%16,
// VGPR v holds K = 2v + 16g (+pos).  Tile index = kt*16 + ntile.
// Flat: ((kt*16+ntile)*32 + lane)*16 + j
// ---------------------------------------------------------------------------
__global__ __launch_bounds__(256) void ldconv_packb(
    const float* __restrict__ wconv, unsigned short* __restrict__ bpack_us)
{
    __bf16* bpack = (__bf16*)bpack_us;
    int t = blockIdx.x * 256 + threadIdx.x;          // 0 .. 294911
    int j    = t & 15;
    int lane = (t >> 4) & 31;
    int tile = t >> 9;
    int kt    = tile >> 4;
    int ntile = tile & 15;
    int g = lane >> 4, nl = lane & 15;
    int v = j >> 1, pos = j & 1;
    int K = kt * 32 + 2 * v + 16 * g + pos;
    int N = ntile * 16 + nl;
    int ic = K / NPAR;
    int kn = K - ic * NPAR;
    float val = wconv[(size_t)N * KDIM + ic * NPAR + kn];
    bpack[t] = (__bf16)val;
}

// ---------------------------------------------------------------------------
// Kernel 3: fused gather + bf16 WMMA GEMM + BN + SiLU.
// Grid: (200 m-tiles, 8 batches). Block 256 = 8 waves.
// WG tile: 32 pixels x 256 oc.  Wave (m_sub = wv&1, n_grp = wv>>1):
//   16 pixels x 64 oc = 4 x v_wmma_f32_16x16x32_bf16 tiles, 36 K-steps.
// ---------------------------------------------------------------------------
__global__ __launch_bounds__(256) void ldconv_gemm(
    const float* __restrict__ x, const float* __restrict__ offs,
    const float* __restrict__ p_n, const unsigned short* __restrict__ bpack_us,
    const float* __restrict__ gamma, const float* __restrict__ beta,
    const float* __restrict__ mean,  const float* __restrict__ var,
    float* __restrict__ out)
{
    const v16bf* bp = (const v16bf*)bpack_us;

    __shared__ int    s_idx[32][NPAR][4];
    __shared__ float  s_wt [32][NPAR][4];
    __shared__ __align__(32) __bf16 s_a[2][32][16];

    const int tid  = threadIdx.x;
    const int lane = tid & 31;
    const int wv   = tid >> 5;
    const int m_sub = wv & 1;
    const int n_grp = wv >> 1;

    const int b = blockIdx.y;
    const int pixBase = blockIdx.x * 32;
    const float* xb = x + (size_t)b * IN_C * HW;

    // ---- per-WG bilinear metadata: 32 pixels x 9 taps -> 4 idx + 4 weights
    for (int e = tid; e < 32 * NPAR; e += 256) {
        int m = e / NPAR, n = e - (e / NPAR) * NPAR;
        int pix = pixBase + m;
        int h = pix / W, w = pix - h * W;
        float ox = offs[((size_t)(b * 18 + n)) * HW + pix];
        float oy = offs[((size_t)(b * 18 + NPAR + n)) * HW + pix];
        float px = (float)h + p_n[n] + ox;
        float py = (float)w + p_n[NPAR + n] + oy;
        float pcx = fminf(fmaxf(px, 0.f), (float)(H - 1));
        float pcy = fminf(fmaxf(py, 0.f), (float)(W - 1));
        float fx = floorf(px), fy = floorf(py);
        int ltx = (int)fminf(fmaxf(fx,       0.f), (float)(H - 1));
        int lty = (int)fminf(fmaxf(fy,       0.f), (float)(W - 1));
        int rbx = (int)fminf(fmaxf(fx + 1.f, 0.f), (float)(H - 1));
        int rby = (int)fminf(fmaxf(fy + 1.f, 0.f), (float)(W - 1));
        float dltx = 1.f + ((float)ltx - pcx);
        float dlty = 1.f + ((float)lty - pcy);
        float drbx = 1.f - ((float)rbx - pcx);
        float drby = 1.f - ((float)rby - pcy);
        s_idx[m][n][0] = ltx * W + lty;
        s_idx[m][n][1] = rbx * W + rby;
        s_idx[m][n][2] = ltx * W + rby;
        s_idx[m][n][3] = rbx * W + lty;
        s_wt[m][n][0] = dltx * dlty;
        s_wt[m][n][1] = drbx * drby;
        s_wt[m][n][2] = dltx * drby;
        s_wt[m][n][3] = drbx * dlty;
    }
    __syncthreads();

    v8f acc[4] = {v8f{}, v8f{}, v8f{}, v8f{}};

    // Each thread stages 4 bf16 A elements per K-tile:
    // flat element id e0 = tid*4 over [s(2)][lane(32)][j(16)]
    const int e0    = tid * 4;
    const int sA    = e0 >> 9;
    const int laneA = (e0 >> 4) & 31;
    const int jb    = e0 & 15;
    const int gA = laneA >> 4;
    const int mA = (laneA & 15) + (sA << 4);

    for (int kt = 0; kt < KT_STEPS; ++kt) {
        // ---- cooperative A-fragment staging (contiguous k = 16g + j order, same as B pack)
        #pragma unroll
        for (int jj = 0; jj < 4; ++jj) {
            int j = jb + jj;
            int v = j >> 1, pos = j & 1;
            int kk = 2 * v + 16 * gA + pos;
            int K = kt * 32 + kk;
            int ic = K / NPAR;
            int n  = K - ic * NPAR;
            const float* xc = xb + (size_t)ic * HW;
            float val = s_wt[mA][n][0] * xc[s_idx[mA][n][0]]
                      + s_wt[mA][n][1] * xc[s_idx[mA][n][1]]
                      + s_wt[mA][n][2] * xc[s_idx[mA][n][2]]
                      + s_wt[mA][n][3] * xc[s_idx[mA][n][3]];
            s_a[sA][laneA][j] = (__bf16)val;
        }
        __syncthreads();

        v16bf afrag = *(const v16bf*)(&s_a[m_sub][lane][0]);
        const v16bf* bbase = bp + (size_t)(kt * 16 + n_grp * 4) * 32 + lane;
        v16bf b0 = bbase[0];
        v16bf b1 = bbase[32];
        v16bf b2 = bbase[64];
        v16bf b3 = bbase[96];

        acc[0] = __builtin_amdgcn_wmma_f32_16x16x32_bf16(false, afrag, false, b0,
                                                         (short)0, acc[0], false, false);
        acc[1] = __builtin_amdgcn_wmma_f32_16x16x32_bf16(false, afrag, false, b1,
                                                         (short)0, acc[1], false, false);
        acc[2] = __builtin_amdgcn_wmma_f32_16x16x32_bf16(false, afrag, false, b2,
                                                         (short)0, acc[2], false, false);
        acc[3] = __builtin_amdgcn_wmma_f32_16x16x32_bf16(false, afrag, false, b3,
                                                         (short)0, acc[3], false, false);
        __syncthreads();
    }

    // ---- fused BN + SiLU epilogue.
    // C layout: lane g=lane/16: VGPR r -> M = r + 8g (pixel row), N = lane%16 (oc)
    const int gC = lane >> 4;
    #pragma unroll
    for (int nt = 0; nt < 4; ++nt) {
        int oc = n_grp * 64 + nt * 16 + (lane & 15);
        float sc = gamma[oc] * rsqrtf(var[oc] + BN_EPS);
        float mn = mean[oc];
        float bt = beta[oc];
        float* op = out + ((size_t)(b * OUT_C + oc)) * HW
                        + pixBase + m_sub * 16 + 8 * gC;
        #pragma unroll
        for (int r = 0; r < 8; ++r) {
            float y = acc[nt][r];
            float z = (y - mn) * sc + bt;
            op[r] = z / (1.f + __expf(-z));
        }
    }
}

// ---------------------------------------------------------------------------
extern "C" void kernel_launch(void* const* d_in, const int* in_sizes, int n_in,
                              void* d_out, int out_size, void* d_ws, size_t ws_size,
                              hipStream_t stream) {
    const float* x       = (const float*)d_in[0];
    const float* w_pconv = (const float*)d_in[1];
    const float* b_pconv = (const float*)d_in[2];
    const float* w_conv  = (const float*)d_in[3];
    const float* bn_g    = (const float*)d_in[4];
    const float* bn_b    = (const float*)d_in[5];
    const float* bn_m    = (const float*)d_in[6];
    const float* bn_v    = (const float*)d_in[7];
    const float* p_n     = (const float*)d_in[8];

    char* ws = (char*)d_ws;
    float* offs = (float*)ws;                                   // 8*18*6400*4 = 3,686,400 B
    unsigned short* bpack = (unsigned short*)(ws + 3686400);    // 1152*256*2  =   589,824 B

    // 1) offsets: 8*18*6400 = 921600 threads
    ldconv_offset<<<3600, 256, 0, stream>>>(x, w_pconv, b_pconv, offs);

    // 2) pack B: 1152*256 = 294912 elements
    ldconv_packb<<<1152, 256, 0, stream>>>(w_conv, bpack);

    // 3) fused gather + WMMA GEMM + BN + SiLU
    dim3 grid(200, 8);
    ldconv_gemm<<<grid, 256, 0, stream>>>(x, offs, p_n, bpack,
                                          bn_g, bn_b, bn_m, bn_v,
                                          (float*)d_out);
}
